// HybridGNN_17428977287455
// MI455X (gfx1250) — matrix operand order repaired and checked by reading, not compiled, not run on hardware
//
#include <hip/hip_runtime.h>

// ---------------------------------------------------------------------------
// HybridGNN for MI455X (gfx1250, wave32).
//  - Edge aggregation: float4 gathers + global_atomic_add_f32 scatters.
//    Working set (100k x 64 f32 = 25.6MB) lives in the 192MB L2; this phase
//    is L2-atomic bound and dominates runtime.
//  - Dense layers: v_wmma_f32_16x16x32_bf16 (f32 accumulate), one wave per
//    16-row node tile, A and B fragments share contiguous k order. Weights are
//    pre-packed once per call into transposed bf16 so each B fragment is a
//    single aligned 32-byte contiguous load (no per-element cvt).
// ---------------------------------------------------------------------------

#define IN_CH 64
#define HID   64
#define HID2  32

typedef __bf16 v16bf __attribute__((ext_vector_type(16)));
typedef float  v8f   __attribute__((ext_vector_type(8)));

// ---------------- zero-fill (graph-capture-safe alternative to memset) -----
__global__ __launch_bounds__(256) void zero_kernel(float4* __restrict__ p,
                                                   size_t n4) {
  size_t i = (size_t)blockIdx.x * blockDim.x + threadIdx.x;
  if (i < n4) p[i] = make_float4(0.f, 0.f, 0.f, 0.f);
}

// ---------------- weight pre-pack: W[CIN][COUT] f32 -> Wt[COUT][CIN] bf16 --
__global__ __launch_bounds__(256) void pack_w_kernel(
    const float* __restrict__ W, __bf16* __restrict__ Wt, int cin, int cout) {
  int idx = blockIdx.x * blockDim.x + threadIdx.x;
  if (idx >= cin * cout) return;
  int k = idx / cout;
  int n = idx % cout;
  Wt[(size_t)n * cin + k] = (__bf16)W[idx];
}

// ---------------- edge scatter: agg[dst] += feat[src], cnt[dst] += 1 -------
template <int CH, bool COUNT>
__global__ __launch_bounds__(256) void scatter_kernel(
    const float* __restrict__ feat, const int* __restrict__ src,
    const int* __restrict__ dst, float* __restrict__ agg,
    float* __restrict__ cnt, int n_edges) {
  constexpr int CHUNKS = CH / 4;
  size_t tid = (size_t)blockIdx.x * blockDim.x + threadIdx.x;
  if (tid >= (size_t)n_edges * CHUNKS) return;
  int e = (int)(tid / CHUNKS);
  int c = (int)(tid % CHUNKS) * 4;      // lanes of a wave: same edge rows,
  int s = src[e];                       // consecutive channels -> coalesced
  int d = dst[e];
  float4 v = *(const float4*)(feat + (size_t)s * CH + c);
  float* ap = agg + (size_t)d * CH + c;
  atomicAdd(ap + 0, v.x);
  atomicAdd(ap + 1, v.y);
  atomicAdd(ap + 2, v.z);
  atomicAdd(ap + 3, v.w);
  if (COUNT && c == 0) atomicAdd(cnt + d, 1.0f);
}

// ---------------- WMMA fragment loaders (same contiguous k order in A and B) ---------
// A: 16x32 bf16 (MxK). lane holds row (lane&15); elem e -> K =
// (lane>>4)*16 + e. `base` points at row start of kTile.
__device__ __forceinline__ v16bf load_a_frag(const float* __restrict__ base,
                                             float scale, int lane) {
  const int kbase = (lane >> 4) * 16;
  v16bf a;
#pragma unroll
  for (int e = 0; e < 16; ++e) {
    int k = kbase + e;
    a[e] = (__bf16)(base[k] * scale);
  }
  return a;
}

// B: 32x16 (KxN) from transposed-packed Wt[COUT][CIN] bf16. lane holds
// column N = nTile*16+(lane&15); elems e are K = kTile*32 + (lane>>4)*16 + e,
// contiguous in Wt -> one aligned 32-byte load per fragment.
template <int CIN>
__device__ __forceinline__ v16bf load_b_frag(const __bf16* __restrict__ Wt,
                                             int kTile, int nTile, int lane) {
  const __bf16* p = Wt + (size_t)(nTile * 16 + (lane & 15)) * CIN +
                    kTile * 32 + (lane >> 4) * 16;
  return *(const v16bf*)p;
}

// ---------------- fused SAGE layer / linear ---------------------------------
// out = opt_relu( (agg/max(cnt,1)) @ Wl + hs @ Wr + bias )
// One wave per 16-row tile; 8 waves (256 threads) per block.
template <int CIN, int COUT, bool HAS_MEAN, bool RELU>
__global__ __launch_bounds__(256) void sage_gemm_kernel(
    const float* __restrict__ agg, const float* __restrict__ cnt,
    const float* __restrict__ hs, const __bf16* __restrict__ WlT,
    const __bf16* __restrict__ WrT, const float* __restrict__ bias,
    float* __restrict__ out, int n_nodes) {
  static_assert(CIN % 32 == 0 && COUT % 16 == 0, "tile shapes");
  constexpr int KT = CIN / 32;
  constexpr int NT = COUT / 16;
  const int lane = threadIdx.x & 31;
  const int wave = threadIdx.x >> 5;
  const int tile = blockIdx.x * 8 + wave;
  const int row0 = tile * 16;
  if (row0 >= n_nodes) return;               // wave-uniform exit
  const bool full = (row0 + 16 <= n_nodes);  // wave-uniform: EXEC stays 1s
  int mrow = row0 + (lane & 15);
  if (mrow >= n_nodes) mrow = n_nodes - 1;   // clamp loads on tail tile

  // Self-features A fragments (reused across all N tiles).
  v16bf aS[KT];
#pragma unroll
  for (int kt = 0; kt < KT; ++kt)
    aS[kt] = load_a_frag(hs + (size_t)mrow * CIN + kt * 32, 1.0f, lane);

  // Neighbor-mean A fragments: fold 1/max(cnt,1) into the bf16 convert.
  v16bf aM[HAS_MEAN ? KT : 1];
  if (HAS_MEAN) {
    float inv = 1.0f / fmaxf(cnt[mrow], 1.0f);
#pragma unroll
    for (int kt = 0; kt < KT; ++kt)
      aM[kt] = load_a_frag(agg + (size_t)mrow * CIN + kt * 32, inv, lane);
  }

#pragma unroll
  for (int nt = 0; nt < NT; ++nt) {
    // Every C/D element this lane owns has N = nt*16 + (lane&15).
    float bv = bias[nt * 16 + (lane & 15)];
    v8f acc;
#pragma unroll
    for (int r = 0; r < 8; ++r) acc[r] = bv;

#pragma unroll
    for (int kt = 0; kt < KT; ++kt) {
      if (HAS_MEAN)
        acc = __builtin_amdgcn_wmma_f32_16x16x32_bf16(
            false, aM[kt], false, load_b_frag<CIN>(WlT, kt, nt, lane),
            (short)0, acc, false, false);
      acc = __builtin_amdgcn_wmma_f32_16x16x32_bf16(
          false, aS[kt], false, load_b_frag<CIN>(WrT, kt, nt, lane),
          (short)0, acc, false, false);
    }

    // C/D layout: VGPR r, this lane -> M = row0 + (lane>>4)*8 + r, fixed N.
    float* orow = out + (size_t)(row0 + (lane >> 4) * 8) * COUT +
                  nt * 16 + (lane & 15);
    if (full) {
#pragma unroll
      for (int r = 0; r < 8; ++r)
        orow[(size_t)r * COUT] = RELU ? fmaxf(acc[r], 0.0f) : acc[r];
    } else {
      const int mbase = row0 + (lane >> 4) * 8;
#pragma unroll
      for (int r = 0; r < 8; ++r)
        if (mbase + r < n_nodes)
          orow[(size_t)r * COUT] = RELU ? fmaxf(acc[r], 0.0f) : acc[r];
    }
  }
}

// ---------------- classifier head: logits = z @ wc + bc (N_CLS = 2) --------
__global__ __launch_bounds__(256) void clf_kernel(
    const float* __restrict__ z, const float* __restrict__ wc,
    const float* __restrict__ bc, float* __restrict__ logits, int n_nodes) {
  int i = blockIdx.x * blockDim.x + threadIdx.x;
  if (i >= n_nodes) return;
  float a0 = bc[0], a1 = bc[1];
  const float* zr = z + (size_t)i * HID2;
#pragma unroll
  for (int k = 0; k < HID2; ++k) {
    float zv = zr[k];
    a0 = fmaf(zv, wc[k * 2 + 0], a0);
    a1 = fmaf(zv, wc[k * 2 + 1], a1);
  }
  logits[(size_t)i * 2 + 0] = a0;
  logits[(size_t)i * 2 + 1] = a1;
}

// ---------------------------------------------------------------------------
extern "C" void kernel_launch(void* const* d_in, const int* in_sizes, int n_in,
                              void* d_out, int out_size, void* d_ws, size_t ws_size,
                              hipStream_t stream) {
  const float* x   = (const float*)d_in[0];
  const int*   ei  = (const int*)d_in[1];
  const float* w1l = (const float*)d_in[2];
  const float* b1  = (const float*)d_in[3];
  const float* w1r = (const float*)d_in[4];
  const float* w2l = (const float*)d_in[5];
  const float* b2  = (const float*)d_in[6];
  const float* w2r = (const float*)d_in[7];
  const float* wp  = (const float*)d_in[8];
  const float* bp  = (const float*)d_in[9];
  const float* wc  = (const float*)d_in[10];
  const float* bc  = (const float*)d_in[11];

  const int n_nodes = in_sizes[0] / IN_CH;
  const int n_edges = in_sizes[1] / 2;
  const int* src = ei;
  const int* dst = ei + n_edges;

  // Workspace: agg[N*64] | cnt[N] | h1[N*64] | h2[N*32] f32, then packed
  // bf16 weights (all offsets keep 32B alignment for v16bf loads).
  float* agg = (float*)d_ws;
  float* cnt = agg + (size_t)n_nodes * 64;
  float* h1  = cnt + n_nodes;
  float* h2  = h1 + (size_t)n_nodes * 64;
  __bf16* w1lT = (__bf16*)(h2 + (size_t)n_nodes * 32);
  __bf16* w1rT = w1lT + 64 * 64;
  __bf16* w2lT = w1rT + 64 * 64;
  __bf16* w2rT = w2lT + 64 * 32;
  __bf16* wpT  = w2rT + 64 * 32;

  // d_out = [logits N*2 | z N*32] flat, in reference return order.
  float* logits = (float*)d_out;
  float* z      = logits + (size_t)n_nodes * 2;

  const int tiles       = (n_nodes + 15) / 16;
  const int gemm_blocks = (tiles + 7) / 8;

  // ---- pack weights (tiny; overlapped with nothing, ~20KB total) ----
  pack_w_kernel<<<(64 * 64 + 255) / 256, 256, 0, stream>>>(w1l, w1lT, 64, 64);
  pack_w_kernel<<<(64 * 64 + 255) / 256, 256, 0, stream>>>(w1r, w1rT, 64, 64);
  pack_w_kernel<<<(64 * 32 + 255) / 256, 256, 0, stream>>>(w2l, w2lT, 64, 32);
  pack_w_kernel<<<(64 * 32 + 255) / 256, 256, 0, stream>>>(w2r, w2rT, 64, 32);
  pack_w_kernel<<<(32 * 32 + 255) / 256, 256, 0, stream>>>(wp, wpT, 32, 32);

  // ---- Layer 1 ----
  {  // zero agg + cnt (contiguous N*65 floats)
    size_t n4 = (size_t)n_nodes * 65 / 4;  // N*65 divisible by 4 (N%4==0)
    zero_kernel<<<(int)((n4 + 255) / 256), 256, 0, stream>>>((float4*)agg, n4);
  }
  {
    size_t total = (size_t)n_edges * (IN_CH / 4);
    scatter_kernel<IN_CH, true><<<(int)((total + 255) / 256), 256, 0, stream>>>(
        x, src, dst, agg, cnt, n_edges);
  }
  sage_gemm_kernel<IN_CH, HID, true, true><<<gemm_blocks, 256, 0, stream>>>(
      agg, cnt, x, w1lT, w1rT, b1, h1, n_nodes);

  // ---- Layer 2 (counts unchanged: same graph) ----
  {
    size_t n4 = (size_t)n_nodes * 64 / 4;
    zero_kernel<<<(int)((n4 + 255) / 256), 256, 0, stream>>>((float4*)agg, n4);
  }
  {
    size_t total = (size_t)n_edges * (HID / 4);
    scatter_kernel<HID, false><<<(int)((total + 255) / 256), 256, 0, stream>>>(
        h1, src, dst, agg, cnt, n_edges);
  }
  sage_gemm_kernel<HID, HID2, true, true><<<gemm_blocks, 256, 0, stream>>>(
      agg, cnt, h1, w2lT, w2rT, b2, h2, n_nodes);

  // ---- proj: z = h2 @ wp + bp (single matmul path, no relu) ----
  sage_gemm_kernel<HID2, HID2, false, false><<<gemm_blocks, 256, 0, stream>>>(
      nullptr, nullptr, h2, nullptr, wpT, bp, z, n_nodes);

  // ---- clf: logits = z @ wc + bc ----
  clf_kernel<<<(n_nodes + 255) / 256, 256, 0, stream>>>(z, wc, bc, logits, n_nodes);
}
